// HGT_59854664237648
// MI455X (gfx1250) — matrix operand order repaired and checked by reading, not compiled, not run on hardware
//
#include <hip/hip_runtime.h>

typedef __bf16 bf16x16 __attribute__((ext_vector_type(16)));
typedef __bf16 bf16x8  __attribute__((ext_vector_type(8)));
typedef float  f32x8   __attribute__((ext_vector_type(8)));
typedef unsigned short u16x8 __attribute__((ext_vector_type(8)));
typedef unsigned int   u32x4 __attribute__((ext_vector_type(4)));
typedef int            i32x8 __attribute__((ext_vector_type(8)));
typedef int            i32x4 __attribute__((ext_vector_type(4)));

#if defined(__has_builtin)
#if __has_builtin(__builtin_amdgcn_tensor_load_to_lds) && \
    __has_builtin(__builtin_amdgcn_s_wait_tensorcnt)
#define HGT_USE_TDM 1
#endif
#endif

namespace {
constexpr int NA = 50000, NP = 100000, NTOT = 150000;
constexpr int HH = 8, LL = 2, NE = 150000;
}

// ---------- scalar bf16 helpers (bit-level, no __bf16 arithmetic) ----------
__device__ __forceinline__ float b2f(unsigned short b) {
  return __uint_as_float(((unsigned)b) << 16);
}
__device__ __forceinline__ unsigned short f2b(float f) {
  unsigned u = __float_as_uint(f);
  unsigned r = u + 0x7FFFu + ((u >> 16) & 1u);   // round-to-nearest-even
  return (unsigned short)(r >> 16);
}
// order-preserving float<->uint encoding for atomicMax on floats
__device__ __forceinline__ unsigned f2ord(float f) {
  unsigned u = __float_as_uint(f);
  return (u & 0x80000000u) ? ~u : (u | 0x80000000u);
}
__device__ __forceinline__ float ord2f(unsigned o) {
  unsigned u = (o & 0x80000000u) ? (o ^ 0x80000000u) : ~o;
  return __uint_as_float(u);
}

// ---------------------------------------------------------------------------
// WMMA GEMM: Y[M, NT*16] = epilogue(A[M,128] (bf16) @ Bpacked[128, NT*16] + bias)
// 256-thread block = 8 waves; each wave owns one 16-row strip.
// The packed B panel (NT*4 KB) is staged once per block into LDS — via the
// Tensor Data Mover (TENSOR_LOAD_TO_LDS + s_wait_tensorcnt) when available,
// else a cooperative copy — then the WMMA inner loop is fed by ds_load_b128.
// EPI: 0 = +bias, store f32 (stride NT*16)
//      1 = +bias, relu, store bf16 (stride 128)
//      2 = +bias, store bf16 (stride 128)
// Bpacked layout: P[((c*NT+n)*32 + lane)*16 + e] = W[c*32 + 16*(lane/16) + e][n*16 + lane%16]
// ---------------------------------------------------------------------------
template <int NT, int EPI>
__global__ __launch_bounds__(256)
void gemm_bf16(const unsigned short* __restrict__ A,
               const unsigned short* __restrict__ Bp,
               const float* __restrict__ bias,
               float* __restrict__ outF,
               unsigned short* __restrict__ outB,
               int M)
{
  constexpr int N = NT * 16;
  __shared__ __align__(32) unsigned short Bs[NT * 2048];

  const int wid  = threadIdx.x >> 5;
  const int lane = threadIdx.x & 31;

#ifdef HGT_USE_TDM
  if (wid == 0) {
    // D# for a 1-row 2D tensor of NT*4096 bytes, global -> LDS (ISA 08 §8)
    const unsigned long long ga = (unsigned long long)Bp;
    const unsigned nbytes = (unsigned)NT * 4096u;
    u32x4 g0;
    g0[0] = 1u;                                            // count=1 (valid), no gather
    g0[1] = (unsigned)(unsigned long long)(void*)Bs;       // lds_addr (low 32b of generic ptr)
    g0[2] = (unsigned)ga;                                  // global_addr[31:0]
    g0[3] = (unsigned)((ga >> 32) & 0x1FFFFFFu) | (2u << 30); // global_addr[56:32] | type=2
    i32x8 g1;
    g1[0] = 0;                                             // wg_mask=0, data_size=1B
    g1[1] = (int)((nbytes & 0xFFFFu) << 16);               // tensor_dim0[15:0]
    g1[2] = (int)((nbytes >> 16) & 0xFFFFu) | (1 << 16);   // tensor_dim0[31:16] | tensor_dim1=1
    g1[3] = (int)((nbytes & 0xFFFFu) << 16);               // tile_dim0 = nbytes
    g1[4] = 1;                                             // tile_dim1 = 1, tile_dim2 = 0
    g1[5] = (int)nbytes;                                   // tensor_dim0_stride
    g1[6] = 0;
    g1[7] = 0;
    i32x4 gz;
    gz[0] = gz[1] = gz[2] = gz[3] = 0;
    i32x8 gz8;
    for (int i = 0; i < 8; ++i) gz8[i] = 0;
    __builtin_amdgcn_tensor_load_to_lds(g0, g1, gz, gz, gz8, 0);
    __builtin_amdgcn_s_wait_tensorcnt(0);
  }
  __syncthreads();
#else
  for (int i = threadIdx.x; i < NT * 256; i += 256) {
    *reinterpret_cast<u16x8*>(Bs + i * 8) =
        *reinterpret_cast<const u16x8*>(Bp + i * 8);
  }
  __syncthreads();
#endif

  const int strip = blockIdx.x * 8 + wid;
  const int rowBase = strip * 16;
  if (rowBase < M) {                            // wave-uniform guard (EXEC all-ones inside)
    const int laneLo = lane & 15, laneHi = lane >> 4;
    const unsigned short* arow = A + (size_t)(rowBase + laneLo) * 128;

    f32x8 acc[NT];
    f32x8 zero;
#pragma unroll
    for (int i = 0; i < 8; ++i) zero[i] = 0.0f;
#pragma unroll
    for (int n = 0; n < NT; ++n) acc[n] = zero;

#pragma unroll
    for (int c = 0; c < 4; ++c) {
      // A 16x32 layout (matches packed B): lanes 0-15 hold K = c*32 + [0..15],
      // lanes 16-31 hold K = c*32 + [16..31]  (two 16B contiguous loads)
      const bf16x8 a0 = *reinterpret_cast<const bf16x8*>(arow + c * 32 + 16 * laneHi);
      const bf16x8 a1 = *reinterpret_cast<const bf16x8*>(arow + c * 32 + 8 + 16 * laneHi);
      const bf16x16 av = __builtin_shufflevector(a0, a1,
          0, 1, 2, 3, 4, 5, 6, 7, 8, 9, 10, 11, 12, 13, 14, 15);
#pragma unroll
      for (int n = 0; n < NT; ++n) {
        const bf16x16 bv = *reinterpret_cast<const bf16x16*>(
            Bs + ((size_t)(c * NT + n) * 32 + lane) * 16);
        acc[n] = __builtin_amdgcn_wmma_f32_16x16x32_bf16(
            false, av, false, bv, (short)0, acc[n], false, false);
      }
    }

    // C/D layout: VGPR r -> row rowBase + r + 8*(lane/16), col = n*16 + lane%16
#pragma unroll
    for (int n = 0; n < NT; ++n) {
      const int col = n * 16 + laneLo;
      const float bv = bias ? bias[col] : 0.0f;
#pragma unroll
      for (int r = 0; r < 8; ++r) {
        const int row = rowBase + r + 8 * laneHi;
        float v = acc[n][r] + bv;
        if (EPI == 0) {
          outF[(size_t)row * N + col] = v;
        } else if (EPI == 1) {
          v = v > 0.0f ? v : 0.0f;
          outB[(size_t)row * 128 + col] = f2b(v);
        } else {
          outB[(size_t)row * 128 + col] = f2b(v);
        }
      }
    }
  }
}

// ------------------------------ small kernels ------------------------------
__global__ void k_cvt_bf16(const float* __restrict__ in,
                           unsigned short* __restrict__ out, size_t n) {
  size_t i = (size_t)blockIdx.x * blockDim.x + threadIdx.x;
  if (i < n) out[i] = f2b(in[i]);
}
__global__ void k_fill_f32(float* __restrict__ p, float v, size_t n) {
  size_t i = (size_t)blockIdx.x * blockDim.x + threadIdx.x;
  if (i < n) p[i] = v;
}
__global__ void k_fill_u32(unsigned* __restrict__ p, unsigned v, size_t n) {
  size_t i = (size_t)blockIdx.x * blockDim.x + threadIdx.x;
  if (i < n) p[i] = v;
}
// Weff[k][h*16+j] = sum_d W[k][h*16+d] * R[h][d][j]   (fold rel matrix into weights)
__global__ void k_eff_w(const float* __restrict__ W, const float* __restrict__ R,
                        float* __restrict__ Weff) {
  int idx = blockIdx.x * blockDim.x + threadIdx.x;
  if (idx >= 128 * 128) return;
  int k = idx >> 7, col = idx & 127, h = col >> 4, j = col & 15;
  float s = 0.0f;
  for (int d = 0; d < 16; ++d) s += W[k * 128 + h * 16 + d] * R[h * 256 + d * 16 + j];
  Weff[idx] = s;
}
__global__ void k_eff_b(const float* __restrict__ b, const float* __restrict__ R,
                        float* __restrict__ beff) {
  int col = threadIdx.x;   // 128 threads
  int h = col >> 4, j = col & 15;
  float s = 0.0f;
  for (int d = 0; d < 16; ++d) s += b[h * 16 + d] * R[h * 256 + d * 16 + j];
  beff[col] = s;
}
// repack f32 W[128, NT*16] into WMMA-B bf16 layout (see gemm_bf16 comment)
__global__ void k_pack_b(const float* __restrict__ W,
                         unsigned short* __restrict__ P, int NT) {
  int idx = blockIdx.x * blockDim.x + threadIdx.x;
  if (idx >= 4 * NT * 512) return;
  int e = idx & 15, lane = (idx >> 4) & 31, n = (idx >> 9) % NT, c = idx / (512 * NT);
  int k = c * 32 + 16 * (lane >> 4) + e;
  int col = n * 16 + (lane & 15);
  P[idx] = f2b(W[k * (NT * 16) + col]);
}

// ------------------------------ edge kernels -------------------------------
__global__ void k_alpha(const unsigned short* __restrict__ Q,
                        const unsigned short* __restrict__ KN,
                        const int* __restrict__ src, const int* __restrict__ dst,
                        const float* __restrict__ relp, float scale,
                        float* __restrict__ alpha, unsigned* __restrict__ mbuf) {
  int tid = blockIdx.x * blockDim.x + threadIdx.x;
  if (tid >= NE * HH) return;
  int e = tid >> 3, h = tid & 7;
  int s = src[e], d = dst[e];
  const unsigned short* q  = Q  + (size_t)d * 128 + h * 16;
  const unsigned short* kk = KN + (size_t)s * 128 + h * 16;
  float acc = 0.0f;
#pragma unroll
  for (int i = 0; i < 16; ++i) acc += b2f(q[i]) * b2f(kk[i]);
  float a = acc * relp[h] * scale;
  alpha[tid] = a;
  atomicMax(mbuf + (size_t)d * HH + h, f2ord(a));
}
__global__ void k_expsum(float* __restrict__ alpha, const int* __restrict__ dst,
                         const unsigned* __restrict__ mbuf, float* __restrict__ sbuf) {
  int tid = blockIdx.x * blockDim.x + threadIdx.x;
  if (tid >= NE * HH) return;
  int e = tid >> 3, h = tid & 7;
  int d = dst[e];
  float mm = ord2f(mbuf[(size_t)d * HH + h]);
  float ex = expf(alpha[tid] - mm);
  alpha[tid] = ex;
  atomicAdd(sbuf + (size_t)d * HH + h, ex);
}
__global__ void k_agg(const float* __restrict__ alpha,
                      const unsigned short* __restrict__ VN,
                      const int* __restrict__ src, const int* __restrict__ dst,
                      const float* __restrict__ sbuf, float* __restrict__ agg) {
  int tid = blockIdx.x * blockDim.x + threadIdx.x;
  if (tid >= NE * HH) return;
  int e = tid >> 3, h = tid & 7;
  int s = src[e], d = dst[e];
  float coef = alpha[tid] / (sbuf[(size_t)d * HH + h] + 1e-16f);
  const unsigned short* v = VN + (size_t)s * 128 + h * 16;
  float* o = agg + (size_t)d * 128 + h * 16;
#pragma unroll
  for (int i = 0; i < 16; ++i) atomicAdd(o + i, coef * b2f(v[i]));
}
__global__ void k_gelu(const float* __restrict__ in,
                       unsigned short* __restrict__ out, size_t n) {
  size_t i = (size_t)blockIdx.x * blockDim.x + threadIdx.x;
  if (i >= n) return;
  float x = in[i];
  out[i] = f2b(0.5f * x * (1.0f + erff(x * 0.70710678118654752f)));
}
__global__ void k_mix(const float* __restrict__ ge,
                      const unsigned short* __restrict__ xold,
                      const float* __restrict__ skipp,
                      unsigned short* __restrict__ xnew, size_t n) {
  size_t i = (size_t)blockIdx.x * blockDim.x + threadIdx.x;
  if (i >= n) return;
  float g = 1.0f / (1.0f + expf(-skipp[0]));
  xnew[i] = f2b(g * ge[i] + (1.0f - g) * b2f(xold[i]));
}

// ---------------------------------------------------------------------------
extern "C" void kernel_launch(void* const* d_in, const int* in_sizes, int n_in,
                              void* d_out, int out_size, void* d_ws, size_t ws_size,
                              hipStream_t stream) {
  (void)in_sizes; (void)n_in; (void)out_size; (void)ws_size;
  const float* x_a   = (const float*)d_in[0];
  const float* x_p   = (const float*)d_in[1];
  const float* win_w = (const float*)d_in[2];
  const float* win_b = (const float*)d_in[3];
  const float* kqv_w = (const float*)d_in[4];
  const float* kqv_b = (const float*)d_in[5];
  const float* a_w   = (const float*)d_in[6];
  const float* a_b   = (const float*)d_in[7];
  const float* skip  = (const float*)d_in[8];
  const float* rel_a = (const float*)d_in[9];
  const float* rel_m = (const float*)d_in[10];
  const float* rel_p = (const float*)d_in[11];
  const float* out_w = (const float*)d_in[12];
  const float* out_b = (const float*)d_in[13];
  const int* ews = (const int*)d_in[14];
  const int* ewd = (const int*)d_in[15];
  const int* ers = (const int*)d_in[16];
  const int* erd = (const int*)d_in[17];
  float* out = (float*)d_out;

  // bump allocator on workspace
  char* wptr = (char*)d_ws;
  auto alloc = [&](size_t bytes) -> void* {
    void* p = (void*)wptr;
    wptr += (bytes + 255) & ~(size_t)255;
    return p;
  };
  const size_t NF = (size_t)NTOT * 128;
  unsigned short* xs0  = (unsigned short*)alloc(NF * 2);
  unsigned short* xs1  = (unsigned short*)alloc(NF * 2);
  unsigned short* Qb   = (unsigned short*)alloc(NF * 2);
  unsigned short* Kb   = (unsigned short*)alloc(NF * 2);
  unsigned short* Vb   = (unsigned short*)alloc(NF * 2);
  unsigned short* tmpB = (unsigned short*)alloc(NF * 2);   // cvt input / gelu buf
  float*    agg   = (float*)alloc(NF * 4);                 // also reused as a_lin f32 out
  float*    alpha = (float*)alloc((size_t)NE * HH * 4);
  unsigned* mbuf  = (unsigned*)alloc((size_t)NP * HH * 4);
  float*    sbuf  = (float*)alloc((size_t)NP * HH * 4);
  float*    Wtmp  = (float*)alloc(128 * 128 * 4);
  float*    beff  = (float*)alloc((size_t)LL * 2 * 2 * 128 * 4);
  unsigned short* pw_win = (unsigned short*)alloc(2 * 16384 * 2);
  unsigned short* pw_q   = (unsigned short*)alloc((size_t)LL * 2 * 16384 * 2);
  unsigned short* pw_k   = (unsigned short*)alloc((size_t)LL * 2 * 16384 * 2);
  unsigned short* pw_v   = (unsigned short*)alloc((size_t)LL * 2 * 16384 * 2);
  unsigned short* pw_a   = (unsigned short*)alloc((size_t)LL * 2 * 16384 * 2);
  unsigned short* pw_out = (unsigned short*)alloc(128 * 64 * 2);

  dim3 b256(256);
  auto g1 = [](size_t n) { return dim3((unsigned)((n + 255) / 256)); };
  auto gG = [](int M) { return dim3((unsigned)((M / 16 + 7) / 8)); };

  // ---- weight preparation: fold rel_a/rel_m into K/V weights, repack all to bf16
  for (int t = 0; t < 2; ++t)
    k_pack_b<<<g1(16384), b256, 0, stream>>>(win_w + (size_t)t * 16384,
                                             pw_win + (size_t)t * 16384, 8);
  k_pack_b<<<g1(8192), b256, 0, stream>>>(out_w, pw_out, 4);
  for (int l = 0; l < LL; ++l)
    for (int t = 0; t < 2; ++t) {
      int lt = l * 2 + t;                       // edge type e == source type t
      const float* Ra = rel_a + (size_t)lt * 2048;
      const float* Rm = rel_m + (size_t)lt * 2048;
      k_pack_b<<<g1(16384), b256, 0, stream>>>(kqv_w + ((size_t)lt * 3 + 1) * 16384,
                                               pw_q + (size_t)lt * 16384, 8);
      k_eff_w<<<g1(16384), b256, 0, stream>>>(kqv_w + ((size_t)lt * 3 + 0) * 16384, Ra, Wtmp);
      k_pack_b<<<g1(16384), b256, 0, stream>>>(Wtmp, pw_k + (size_t)lt * 16384, 8);
      k_eff_b<<<1, 128, 0, stream>>>(kqv_b + ((size_t)lt * 3 + 0) * 128, Ra,
                                     beff + ((size_t)lt * 2 + 0) * 128);
      k_eff_w<<<g1(16384), b256, 0, stream>>>(kqv_w + ((size_t)lt * 3 + 2) * 16384, Rm, Wtmp);
      k_pack_b<<<g1(16384), b256, 0, stream>>>(Wtmp, pw_v + (size_t)lt * 16384, 8);
      k_eff_b<<<1, 128, 0, stream>>>(kqv_b + ((size_t)lt * 3 + 2) * 128, Rm,
                                     beff + ((size_t)lt * 2 + 1) * 128);
      k_pack_b<<<g1(16384), b256, 0, stream>>>(a_w + (size_t)lt * 16384,
                                               pw_a + (size_t)lt * 16384, 8);
    }

  // ---- input projection: xs = relu(x @ win_w + win_b), stored bf16
  k_cvt_bf16<<<g1((size_t)NA * 128), b256, 0, stream>>>(x_a, tmpB, (size_t)NA * 128);
  k_cvt_bf16<<<g1((size_t)NP * 128), b256, 0, stream>>>(x_p, tmpB + (size_t)NA * 128,
                                                        (size_t)NP * 128);
  gemm_bf16<8, 1><<<gG(NA), b256, 0, stream>>>(tmpB, pw_win, win_b, nullptr, xs0, NA);
  gemm_bf16<8, 1><<<gG(NP), b256, 0, stream>>>(tmpB + (size_t)NA * 128, pw_win + 16384,
                                               win_b + 128, nullptr,
                                               xs0 + (size_t)NA * 128, NP);

  unsigned short* xs = xs0;
  unsigned short* xsn = xs1;

  for (int l = 0; l < LL; ++l) {
    // Q / kn=K@Ra / vn=V@Rm per node type (rel transform folded into weights)
    for (int t = 0; t < 2; ++t) {
      int lt = l * 2 + t;
      size_t base = (t == 0) ? 0 : (size_t)NA * 128;
      int Mt = (t == 0) ? NA : NP;
      gemm_bf16<8, 2><<<gG(Mt), b256, 0, stream>>>(
          xs + base, pw_q + (size_t)lt * 16384, kqv_b + ((size_t)lt * 3 + 1) * 128,
          nullptr, Qb + base, Mt);
      gemm_bf16<8, 2><<<gG(Mt), b256, 0, stream>>>(
          xs + base, pw_k + (size_t)lt * 16384, beff + ((size_t)lt * 2 + 0) * 128,
          nullptr, Kb + base, Mt);
      gemm_bf16<8, 2><<<gG(Mt), b256, 0, stream>>>(
          xs + base, pw_v + (size_t)lt * 16384, beff + ((size_t)lt * 2 + 1) * 128,
          nullptr, Vb + base, Mt);
    }
    k_fill_f32<<<g1(NF), b256, 0, stream>>>(agg, 0.0f, NF);
    // edge phase: attention + segment softmax + scatter aggregate
    for (int et = 0; et < 2; ++et) {
      int st = et, dt = 1 - et;
      const int* src = et ? ers : ews;
      const int* dst = et ? erd : ewd;
      size_t sbase = st ? (size_t)NA * 128 : 0;
      size_t dbase = dt ? (size_t)NA * 128 : 0;
      int Ndt = dt ? NP : NA;
      k_fill_u32<<<g1((size_t)Ndt * HH), b256, 0, stream>>>(mbuf, 0u, (size_t)Ndt * HH);
      k_fill_f32<<<g1((size_t)Ndt * HH), b256, 0, stream>>>(sbuf, 0.0f, (size_t)Ndt * HH);
      k_alpha<<<g1((size_t)NE * HH), b256, 0, stream>>>(
          Qb + dbase, Kb + sbase, src, dst, rel_p + (size_t)(l * 2 + et) * HH,
          0.25f /* 1/sqrt(16) */, alpha, mbuf);
      k_expsum<<<g1((size_t)NE * HH), b256, 0, stream>>>(alpha, dst, mbuf, sbuf);
      k_agg<<<g1((size_t)NE * HH), b256, 0, stream>>>(alpha, Vb + sbase, src, dst, sbuf,
                                                      agg + dbase);
    }
    // gelu -> a_lin GEMM -> gated skip mix
    for (int t = 0; t < 2; ++t) {
      int lt = l * 2 + t;
      size_t base = (t == 0) ? 0 : (size_t)NA * 128;
      int Mt = (t == 0) ? NA : NP;
      k_gelu<<<g1((size_t)Mt * 128), b256, 0, stream>>>(agg + base, tmpB + base,
                                                        (size_t)Mt * 128);
      gemm_bf16<8, 0><<<gG(Mt), b256, 0, stream>>>(
          tmpB + base, pw_a + (size_t)lt * 16384, a_b + (size_t)lt * 128,
          agg + base, nullptr, Mt);
      k_mix<<<g1((size_t)Mt * 128), b256, 0, stream>>>(agg + base, xs + base, skip + lt,
                                                       xsn + base, (size_t)Mt * 128);
    }
    unsigned short* tsw = xs; xs = xsn; xsn = tsw;
  }

  // final classifier on author nodes: [NA,128] @ [128,64] + b -> f32 d_out
  gemm_bf16<4, 0><<<gG(NA), b256, 0, stream>>>(xs, pw_out, out_b, out, nullptr, NA);
}
